// GPT_89240830476836
// MI455X (gfx1250) — matrix operand order repaired and checked by reading, not compiled, not run on hardware
//
#include <hip/hip_runtime.h>
#include <hip/hip_bf16.h>

// GPT-2 small forward on gfx1250: bf16 WMMA GEMMs + WMMA flash-attention.
#define Lc 6
#define Hh 12
#define Cc 768
#define Vv 50257
#define Bb 4
#define Tt 1024
#define HDd 64
#define FFd (4*Cc)
#define BTd (Bb*Tt)
#define EPSf 1e-5f

typedef __bf16 bf16;
typedef bf16  bf16x8  __attribute__((ext_vector_type(8)));
typedef bf16  bf16x16 __attribute__((ext_vector_type(16)));
typedef float floatx8 __attribute__((ext_vector_type(8)));

// Build a v16bf A/B fragment from two 8-element (16-byte aligned) LDS chunks.
static __device__ __forceinline__ bf16x16 make_frag(const bf16* lo, const bf16* hi) {
  bf16x8 a = *(const bf16x8*)lo;
  bf16x8 b = *(const bf16x8*)hi;
  bf16x16 r;
#pragma unroll
  for (int i = 0; i < 8; ++i) { r[i] = a[i]; r[i + 8] = b[i]; }
  return r;
}

static __device__ __forceinline__ floatx8 wmma_bf16(bf16x16 a, bf16x16 b, floatx8 c) {
  return __builtin_amdgcn_wmma_f32_16x16x32_bf16(false, a, false, b, (short)0, c, false, false);
}

// ---------------------------------------------------------------- embedding
__global__ void embed_kernel(const int* __restrict__ idx, const float* __restrict__ wte,
                             const float* __restrict__ wpe, float* __restrict__ x) {
  int row = blockIdx.x;
  int tok = idx[row];
  int t = row & (Tt - 1);
  for (int c = threadIdx.x; c < Cc; c += blockDim.x)
    x[(size_t)row * Cc + c] = wte[(size_t)tok * Cc + c] + wpe[(size_t)t * Cc + c];
}

// ---------------------------------------------------------------- layernorm
__global__ __launch_bounds__(256) void layernorm_kernel(const float* __restrict__ in,
                                                        float* __restrict__ out,
                                                        const float* __restrict__ g,
                                                        const float* __restrict__ b) {
  __shared__ float red[256];
  const int row = blockIdx.x, tid = threadIdx.x;
  const float* xr = in + (size_t)row * Cc;
  float s = 0.f;
  for (int c = tid; c < Cc; c += 256) s += xr[c];
  red[tid] = s; __syncthreads();
  for (int off = 128; off > 0; off >>= 1) { if (tid < off) red[tid] += red[tid + off]; __syncthreads(); }
  const float mu = red[0] * (1.f / Cc);
  __syncthreads();
  float v = 0.f;
  for (int c = tid; c < Cc; c += 256) { float d = xr[c] - mu; v += d * d; }
  red[tid] = v; __syncthreads();
  for (int off = 128; off > 0; off >>= 1) { if (tid < off) red[tid] += red[tid + off]; __syncthreads(); }
  const float rstd = rsqrtf(red[0] * (1.f / Cc) + EPSf);
  for (int c = tid; c < Cc; c += 256)
    out[(size_t)row * Cc + c] = (xr[c] - mu) * rstd * g[c] + b[c];
}

// ---------------------------------------------------------------- GEMM
// C[M,N] = A[M,K] @ B + bias (+ residual) (+ GELU); A,B fp32 in memory,
// converted to bf16 while staging to LDS. 128x128 block tile, 8 waves,
// each wave owns a 32x64 strip (2x4 accumulators). B layout: k-major [K,N]
// or n-major [N,K] (GF_BNMAJOR, used for logits @ wte^T).
#define GF_BNMAJOR 1
#define GF_GELU    2

__global__ __launch_bounds__(256) void gemm_kernel(const float* __restrict__ A,
                                                   const float* __restrict__ Bg,
                                                   const float* __restrict__ bias,
                                                   const float* __restrict__ res,
                                                   float* __restrict__ Cout,
                                                   int M, int N, int K, int flags) {
  __shared__ bf16 As[128][40];  // padded rows: 80B stride, conflict-free frag reads
  __shared__ bf16 Bs[128][40];  // n-major: Bs[n][k]
  const int tid = threadIdx.x;
  const int lane = tid & 31, w = tid >> 5;
  const int hf = lane >> 4, nl = lane & 15;
  const int wm = w >> 1, wn = w & 1;               // 4x2 wave grid
  const int n0 = blockIdx.x * 128, m0 = blockIdx.y * 128;

  floatx8 acc[2][4];
#pragma unroll
  for (int im = 0; im < 2; ++im)
#pragma unroll
    for (int j = 0; j < 4; ++j)
#pragma unroll
      for (int e = 0; e < 8; ++e) acc[im][j][e] = 0.f;

  const int ksteps = K >> 5;
  for (int kt = 0; kt < ksteps; ++kt) {
    const int k0 = kt << 5;
    __syncthreads();
    for (int i = tid; i < 1024; i += 256) {         // A tile 128x32
      int r = i >> 3, c4 = (i & 7) << 2;
      float4 v = *(const float4*)&A[(size_t)(m0 + r) * K + k0 + c4];
      As[r][c4 + 0] = (bf16)v.x; As[r][c4 + 1] = (bf16)v.y;
      As[r][c4 + 2] = (bf16)v.z; As[r][c4 + 3] = (bf16)v.w;
    }
    if (flags & GF_BNMAJOR) {                       // B global [N,K]
      for (int i = tid; i < 1024; i += 256) {
        int r = i >> 3, c4 = (i & 7) << 2;
        int gn = n0 + r;
        float4 v = make_float4(0.f, 0.f, 0.f, 0.f);
        if (gn < N) v = *(const float4*)&Bg[(size_t)gn * K + k0 + c4];
        Bs[r][c4 + 0] = (bf16)v.x; Bs[r][c4 + 1] = (bf16)v.y;
        Bs[r][c4 + 2] = (bf16)v.z; Bs[r][c4 + 3] = (bf16)v.w;
      }
    } else {                                        // B global [K,N] -> transpose
      for (int i = tid; i < 4096; i += 256) {
        int c = i >> 7, n = i & 127, gn = n0 + n;
        float v = (gn < N) ? Bg[(size_t)(k0 + c) * N + gn] : 0.f;
        Bs[n][c] = (bf16)v;
      }
    }
    // prefetch next K tile into L2 / WGP$ (global_prefetch_b8)
    if (kt + 1 < ksteps) {
      __builtin_prefetch(&A[(size_t)(m0 + (tid >> 1)) * K + k0 + 32 + ((tid & 1) << 4)], 0, 0);
      if (flags & GF_BNMAJOR) {
        int gn = n0 + (tid >> 1);
        if (gn < N) __builtin_prefetch(&Bg[(size_t)gn * K + k0 + 32 + ((tid & 1) << 4)], 0, 0);
      } else {
        int gn = n0 + ((tid & 7) << 4);
        if (gn < N) __builtin_prefetch(&Bg[(size_t)(k0 + 32 + (tid >> 3)) * N + gn], 0, 0);
      }
    }
    __syncthreads();
    // load ALL fragments first, then issue all WMMAs (batch the DS waits)
    bf16x16 af[2], bfr[4];
#pragma unroll
    for (int im = 0; im < 2; ++im) {
      const bf16* ap = &As[wm * 32 + im * 16 + nl][8 * hf];
      af[im] = make_frag(ap, ap + 16);
    }
#pragma unroll
    for (int j = 0; j < 4; ++j) {
      const bf16* bp = &Bs[wn * 64 + j * 16 + nl][8 * hf];
      bfr[j] = make_frag(bp, bp + 16);
    }
#pragma unroll
    for (int im = 0; im < 2; ++im)
#pragma unroll
      for (int j = 0; j < 4; ++j)
        acc[im][j] = wmma_bf16(af[im], bfr[j], acc[im][j]);
  }
  // epilogue: D layout -> row = 8*half + vgpr, col = j*16 + lane%16
#pragma unroll
  for (int im = 0; im < 2; ++im)
#pragma unroll
    for (int j = 0; j < 4; ++j) {
      int gn = n0 + wn * 64 + j * 16 + nl;
      if (gn < N) {
        float bv = bias ? bias[gn] : 0.f;
#pragma unroll
        for (int r = 0; r < 8; ++r) {
          int gm = m0 + wm * 32 + im * 16 + 8 * hf + r;
          float v = acc[im][j][r] + bv;
          if (res) v += res[(size_t)gm * N + gn];
          if (flags & GF_GELU) {
            float x3 = v * v * v;
            v = 0.5f * v * (1.f + tanhf(0.7978845608028654f * (v + 0.044715f * x3)));
          }
          Cout[(size_t)gm * N + gn] = v;
        }
      }
    }
}

// ---------------------------------------------------------------- attention
// One block per (b,h, 64-query tile); flash-attention with WMMA for QK^T and PV.
__global__ __launch_bounds__(128) void attn_kernel(const float* __restrict__ qkv,
                                                   float* __restrict__ y) {
  __shared__ bf16 Qs[64][72];   // [query][d], pre-scaled by 1/sqrt(HD)
  __shared__ bf16 Ks[64][72];   // [key][d]   (n-major for S = Q K^T)
  __shared__ bf16 Vt[64][72];   // [d][key]   (n-major for O += P V)
  __shared__ bf16 Ps[64][72];   // staging to reshape P into A fragments
  const int tid = threadIdx.x;
  const int lane = tid & 31, w = tid >> 5;
  const int hf = lane >> 4, nl = lane & 15;
  const int qb = blockIdx.x, bh = blockIdx.y;
  const int b = bh / Hh, h = bh % Hh;
  const int q0 = qb * 64;
  const size_t base = (size_t)b * Tt * (3 * Cc);
  const int qoff = h * HDd, koff = Cc + h * HDd, voff = 2 * Cc + h * HDd;

  for (int i = tid; i < 64 * 16; i += 128) {       // stage Q once
    int t = i >> 4, d4 = (i & 15) << 2;
    float4 v = *(const float4*)&qkv[base + (size_t)(q0 + t) * (3 * Cc) + qoff + d4];
    Qs[t][d4 + 0] = (bf16)(v.x * 0.125f); Qs[t][d4 + 1] = (bf16)(v.y * 0.125f);
    Qs[t][d4 + 2] = (bf16)(v.z * 0.125f); Qs[t][d4 + 3] = (bf16)(v.w * 0.125f);
  }

  floatx8 o[4];
  float mi[8], li[8];
#pragma unroll
  for (int j = 0; j < 4; ++j)
#pragma unroll
    for (int e = 0; e < 8; ++e) o[j][e] = 0.f;
#pragma unroll
  for (int r = 0; r < 8; ++r) { mi[r] = -3.0e38f; li[r] = 0.f; }

  for (int kb = 0; kb <= qb; ++kb) {
    __syncthreads();
    const int kk0 = kb * 64;
    for (int i = tid; i < 64 * 16; i += 128) {     // stage K row-major, V transposed
      int t = i >> 4, d4 = (i & 15) << 2;
      float4 kv = *(const float4*)&qkv[base + (size_t)(kk0 + t) * (3 * Cc) + koff + d4];
      Ks[t][d4 + 0] = (bf16)kv.x; Ks[t][d4 + 1] = (bf16)kv.y;
      Ks[t][d4 + 2] = (bf16)kv.z; Ks[t][d4 + 3] = (bf16)kv.w;
      float4 vv = *(const float4*)&qkv[base + (size_t)(kk0 + t) * (3 * Cc) + voff + d4];
      Vt[d4 + 0][t] = (bf16)vv.x; Vt[d4 + 1][t] = (bf16)vv.y;
      Vt[d4 + 2][t] = (bf16)vv.z; Vt[d4 + 3][t] = (bf16)vv.w;
    }
    __syncthreads();

    floatx8 s[4];
#pragma unroll
    for (int j = 0; j < 4; ++j)
#pragma unroll
      for (int e = 0; e < 8; ++e) s[j][e] = 0.f;
#pragma unroll
    for (int ks = 0; ks < 2; ++ks) {               // S = Q K^T  (HD=64 -> 2 k-steps)
      const bf16* ap = &Qs[w * 16 + nl][ks * 32 + 8 * hf];
      bf16x16 af = make_frag(ap, ap + 16);
      bf16x16 bfr[4];
#pragma unroll
      for (int j = 0; j < 4; ++j) {
        const bf16* bp = &Ks[j * 16 + nl][ks * 32 + 8 * hf];
        bfr[j] = make_frag(bp, bp + 16);
      }
#pragma unroll
      for (int j = 0; j < 4; ++j) s[j] = wmma_bf16(af, bfr[j], s[j]);
    }
    if (kb == qb) {                                // causal mask on diagonal tile
#pragma unroll
      for (int j = 0; j < 4; ++j) {
        int key = kk0 + j * 16 + nl;
#pragma unroll
        for (int r = 0; r < 8; ++r)
          if (key > q0 + w * 16 + 8 * hf + r) s[j][r] = -3.0e38f;
      }
    }
    // online softmax: per row (vgpr r) reduce across 4 tiles + 16 lanes of half
#pragma unroll
    for (int r = 0; r < 8; ++r) {
      float mx = fmaxf(fmaxf(s[0][r], s[1][r]), fmaxf(s[2][r], s[3][r]));
      mx = fmaxf(mx, __shfl_xor(mx, 1, 32));
      mx = fmaxf(mx, __shfl_xor(mx, 2, 32));
      mx = fmaxf(mx, __shfl_xor(mx, 4, 32));
      mx = fmaxf(mx, __shfl_xor(mx, 8, 32));
      float newm = fmaxf(mi[r], mx);
      float scale = __expf(mi[r] - newm);
      float rsum = 0.f;
#pragma unroll
      for (int j = 0; j < 4; ++j) {
        float p = __expf(s[j][r] - newm);
        s[j][r] = p;
        rsum += p;
      }
      rsum += __shfl_xor(rsum, 1, 32);
      rsum += __shfl_xor(rsum, 2, 32);
      rsum += __shfl_xor(rsum, 4, 32);
      rsum += __shfl_xor(rsum, 8, 32);
      li[r] = li[r] * scale + rsum;
      mi[r] = newm;
#pragma unroll
      for (int j = 0; j < 4; ++j) o[j][r] *= scale;
    }
    // reshape P (C-layout) -> A-fragment layout via LDS (same-wave, DS in-order)
#pragma unroll
    for (int j = 0; j < 4; ++j)
#pragma unroll
      for (int r = 0; r < 8; ++r)
        Ps[w * 16 + 8 * hf + r][j * 16 + nl] = (bf16)s[j][r];
    asm volatile("s_wait_dscnt 0x0" ::: "memory");
#pragma unroll
    for (int ks = 0; ks < 2; ++ks) {               // O += P V
      const bf16* ap = &Ps[w * 16 + nl][ks * 32 + 8 * hf];
      bf16x16 af = make_frag(ap, ap + 16);
      bf16x16 bfr[4];
#pragma unroll
      for (int j = 0; j < 4; ++j) {
        const bf16* bp = &Vt[j * 16 + nl][ks * 32 + 8 * hf];
        bfr[j] = make_frag(bp, bp + 16);
      }
#pragma unroll
      for (int j = 0; j < 4; ++j) o[j] = wmma_bf16(af, bfr[j], o[j]);
    }
  }
  // normalize and scatter back to [B,T,C]
#pragma unroll
  for (int j = 0; j < 4; ++j)
#pragma unroll
    for (int r = 0; r < 8; ++r) {
      int t = q0 + w * 16 + 8 * hf + r;
      y[((size_t)(b * Tt + t)) * Cc + h * HDd + j * 16 + nl] = o[j][r] / li[r];
    }
}

// ---------------------------------------------------------------- driver
extern "C" void kernel_launch(void* const* d_in, const int* in_sizes, int n_in,
                              void* d_out, int out_size, void* d_ws, size_t ws_size,
                              hipStream_t stream) {
  (void)in_sizes; (void)n_in; (void)out_size; (void)ws_size;
  const int*   idx   = (const int*)d_in[0];
  const float* wte   = (const float*)d_in[1];
  const float* wpe   = (const float*)d_in[2];
  const float* ln1s  = (const float*)d_in[3];
  const float* ln1b  = (const float*)d_in[4];
  const float* qkvw  = (const float*)d_in[5];
  const float* qkvb  = (const float*)d_in[6];
  const float* projw = (const float*)d_in[7];
  const float* projb = (const float*)d_in[8];
  const float* ln2s  = (const float*)d_in[9];
  const float* ln2b  = (const float*)d_in[10];
  const float* fcw   = (const float*)d_in[11];
  const float* fcb   = (const float*)d_in[12];
  const float* fcpw  = (const float*)d_in[13];
  const float* fcpb  = (const float*)d_in[14];
  const float* lnfs  = (const float*)d_in[15];
  const float* lnfb  = (const float*)d_in[16];
  float* out = (float*)d_out;

  float* x    = (float*)d_ws;                      // [BT,C]
  float* hbuf = x    + (size_t)BTd * Cc;           // [BT,C]   (LN output)
  float* qkv  = hbuf + (size_t)BTd * Cc;           // [BT,3C]
  float* ybuf = qkv  + (size_t)BTd * 3 * Cc;       // [BT,C]   (attention out)
  float* mlp  = ybuf + (size_t)BTd * Cc;           // [BT,4C]

  embed_kernel<<<BTd, 256, 0, stream>>>(idx, wte, wpe, x);
  for (int l = 0; l < Lc; ++l) {
    layernorm_kernel<<<BTd, 256, 0, stream>>>(x, hbuf, ln1s + l * Cc, ln1b + l * Cc);
    gemm_kernel<<<dim3(3 * Cc / 128, BTd / 128), 256, 0, stream>>>(
        hbuf, qkvw + (size_t)l * Cc * 3 * Cc, qkvb + (size_t)l * 3 * Cc,
        nullptr, qkv, BTd, 3 * Cc, Cc, 0);
    attn_kernel<<<dim3(Tt / 64, Bb * Hh), 128, 0, stream>>>(qkv, ybuf);
    gemm_kernel<<<dim3(Cc / 128, BTd / 128), 256, 0, stream>>>(
        ybuf, projw + (size_t)l * Cc * Cc, projb + (size_t)l * Cc,
        x, x, BTd, Cc, Cc, 0);
    layernorm_kernel<<<BTd, 256, 0, stream>>>(x, hbuf, ln2s + l * Cc, ln2b + l * Cc);
    gemm_kernel<<<dim3(FFd / 128, BTd / 128), 256, 0, stream>>>(
        hbuf, fcw + (size_t)l * Cc * FFd, fcb + (size_t)l * FFd,
        nullptr, mlp, BTd, FFd, Cc, GF_GELU);
    gemm_kernel<<<dim3(Cc / 128, BTd / 128), 256, 0, stream>>>(
        mlp, fcpw + (size_t)l * FFd * Cc, fcpb + (size_t)l * Cc,
        x, x, BTd, Cc, FFd, 0);
  }
  layernorm_kernel<<<BTd, 256, 0, stream>>>(x, hbuf, lnfs, lnfb);
  gemm_kernel<<<dim3((Vv + 127) / 128, BTd / 128), 256, 0, stream>>>(
      hbuf, wte, nullptr, nullptr, out, BTd, Vv, Cc, GF_BNMAJOR);
}
